// DualCNNCMATransformer_62027917689518
// MI455X (gfx1250) — matrix operand order repaired and checked by reading, not compiled, not run on hardware
//
#include <hip/hip_runtime.h>
#include <cstddef>
#include <cstdint>

// ---------------------------------------------------------------------------
// DualCNN CMA Transformer for MI455X (gfx1250, wave32, WMMA + TDM)
// B=32, C=2048, H=W=8 (N=64).
// Roofline: ~1.06 TFLOP, dominated by 1.03 GB of f32 weights streamed once
// (~44 us @ 23.3 TB/s). Weights: f32 -> bf16 in the A-fragment build.
// Activations: bf16 end-to-end. All channel mixes: v_wmma_f32_16x16x32_bf16.
// Activation K-tiles: double-buffered Tensor Data Mover copies into LDS with
// DMA-inserted row padding (stride 66 halves) for bank-conflict-free gathers;
// compute overlaps the in-flight DMA (s_wait_tensorcnt 1).
// ---------------------------------------------------------------------------

#define kB  32
#define kC  2048
#define kN  64      // H*W
#define kH  8
#define kW  8
#define kDK 512     // channels per attention branch
#define kS  66      // LDS row stride in halves: 64 data + 2 DMA pad
#define kCH (32 * kS)   // halves per staged K-chunk buffer (2112)

typedef unsigned short bfu;
typedef __attribute__((ext_vector_type(16))) __bf16 v16bf;
typedef __attribute__((ext_vector_type(8)))  float  v8f;

union BF16Frag { v16bf v; bfu s[16]; };

__device__ __forceinline__ bfu f2bf(float f) {
  union { float f; unsigned u; } c; c.f = f;
  return (bfu)((c.u + 0x7FFFu + ((c.u >> 16) & 1u)) >> 16);  // RNE
}
__device__ __forceinline__ float bf2f(bfu h) {
  union { unsigned u; float f; } c; c.u = ((unsigned)h) << 16; return c.f;
}

// ---------------------------------------------------------------------------
// Tensor Data Mover: copy 2048 contiguous bf16 (one 32x64 K-chunk) to LDS,
// inserting 1 DWORD of LDS padding every 32 DWORDs (=> 66-half row stride).
// D# per CDNA5 ISA §8.3-8.6; 6-arg builtin on this toolchain.
// ---------------------------------------------------------------------------
#if __has_builtin(__builtin_amdgcn_tensor_load_to_lds) && __has_builtin(__builtin_amdgcn_s_wait_tensorcnt)
#define USE_TDM 1
typedef unsigned int v4u __attribute__((ext_vector_type(4)));
typedef int v8i __attribute__((ext_vector_type(8)));
typedef int v4i __attribute__((ext_vector_type(4)));

__device__ __forceinline__ void tdm_copy(unsigned lds_off, const void* g, unsigned n) {
  unsigned long long ga = (unsigned long long)g;
  v4u g0;
  g0[0] = 1u;                                                  // D# count = 1 (user)
  g0[1] = lds_off;                                             // lds_addr
  g0[2] = (unsigned)ga;                                        // global_addr[31:0]
  g0[3] = (unsigned)((ga >> 32) & 0x01FFFFFFu) | (2u << 30);   // addr[56:32] | type=2
  v8i g1;
  g1[0] = (int)((1u << 16) | (1u << 20) | (4u << 22));
  // data_size=1 (2B) | pad_enable | pad_interval=4 (32 DWORDs), pad_amount=0 (1 DWORD)
  g1[1] = (int)(n << 16);                  // tensor_dim0[15:0]   (bits 63:48)
  g1[2] = (int)((n >> 16) | (1u << 16));   // tensor_dim0[31:16] | tensor_dim1=1
  g1[3] = (int)(n << 16);                  // tile_dim0           (bits 127:112)
  g1[4] = 1;                               // tile_dim1=1, tile_dim2=0
  g1[5] = (int)n;                          // tensor_dim0_stride[31:0]
  g1[6] = (int)(n << 16);                  // stride0 hi=0 | tensor_dim1_stride[15:0]
  g1[7] = (int)(n >> 16);                  // tensor_dim1_stride[47:16]
  v4i g2; g2[0] = 1; g2[1] = 1; g2[2] = 0; g2[3] = 0;  // tensor_dim2=1, tensor_dim3=1
  v4i g3; g3[0] = 0; g3[1] = 0; g3[2] = 0; g3[3] = 0;
  v8i g4; g4[0] = 0; g4[1] = 0; g4[2] = 0; g4[3] = 0;
          g4[4] = 0; g4[5] = 0; g4[6] = 0; g4[7] = 0;  // unused extra group
  __builtin_amdgcn_tensor_load_to_lds(g0, g1, g2, g3, g4, 0);
}
#endif

// ---------------------------------------------------------------------------
// conv1x1 as GEMM:  Y[b,o,n] = sum_c W[o,c] * X[b,c,n] + bias[o]
// grid (O/64, B), 128 threads = 4 waves; wave w -> rows [w*16, w*16+16)
// ---------------------------------------------------------------------------
__global__ void __launch_bounds__(128)
gemm1x1(const float* __restrict__ W, const float* __restrict__ bias,
        const bfu* __restrict__ X, bfu* __restrict__ Y, int O) {
  __shared__ __attribute__((aligned(16))) bfu xs[2 * kCH];   // double-buffered K x N
  const int b     = blockIdx.y;
  const int wave  = threadIdx.x >> 5;
  const int lane  = threadIdx.x & 31;
  const int laneM = lane & 15;
  const int hi    = lane >> 4;
  const int row0  = blockIdx.x * 64 + wave * 16;
  const bfu* Xb   = X + (size_t)b * kC * kN;

  v8f acc[4] = {};

#ifdef USE_TDM
  const unsigned lds0 = (unsigned)(unsigned long long)(void*)xs;
  if (wave == 0) tdm_copy(lds0, Xb, 32 * kN);       // prefetch chunk 0
#endif
  for (int k0 = 0; k0 < kC; k0 += 32) {
    const int cur = (k0 >> 5) & 1;
    const bfu* xsc = xs + cur * kCH;
#ifdef USE_TDM
    if (wave == 0) {
      if (k0 + 32 < kC) {                           // prefetch next, overlap compute
        tdm_copy(lds0 + (unsigned)((1 - cur) * kCH * 2),
                 Xb + (size_t)(k0 + 32) * kN, 32 * kN);
        __builtin_amdgcn_s_wait_tensorcnt(1);       // chunk k0 complete (in-order)
      } else {
        __builtin_amdgcn_s_wait_tensorcnt(0);
      }
    }
    __syncthreads();
#else
    __syncthreads();
    for (int i = threadIdx.x * 2; i < 32 * kN; i += 128 * 2) {
      const int kk = i >> 6, nn = i & 63;
      *(unsigned*)(xs + cur * kCH + kk * kS + nn) =
          *(const unsigned*)(Xb + (size_t)k0 * kN + i);
    }
    __syncthreads();
#endif
    // A fragment (16-bit A 16x32 layout): lane->M=laneM; e<8 => K=hi*8+e,
    // e>=8 => K=16+hi*8+(e-8). Weights: the only f32->bf16 conversion.
    BF16Frag a;
    const float* wr = W + (size_t)(row0 + laneM) * kC + k0 + hi * 8;
#pragma unroll
    for (int e = 0; e < 8; ++e) { a.s[e] = f2bf(wr[e]); a.s[8 + e] = f2bf(wr[16 + e]); }
    const int klo = hi * 8;    // B layout: e<8 => K=hi*8+e, e>=8 => K=16+hi*8+(e-8)
#pragma unroll
    for (int t = 0; t < 4; ++t) {
      BF16Frag bm;
      const bfu* col = xsc + klo * kS + t * 16 + laneM;
#pragma unroll
      for (int e = 0; e < 16; ++e) bm.s[e] = col[(e < 8 ? e : e + 8) * kS];
      acc[t] = __builtin_amdgcn_wmma_f32_16x16x32_bf16(false, a.v, false, bm.v,
                                                       (short)0, acc[t], false, false);
    }
    __syncthreads();           // release buffer before DMA overwrites it
  }
  float bv[8];
#pragma unroll
  for (int j = 0; j < 8; ++j) bv[j] = bias[row0 + hi * 8 + j];
  bfu* Yb = Y + (size_t)b * O * kN;
#pragma unroll
  for (int t = 0; t < 4; ++t)
#pragma unroll
    for (int j = 0; j < 8; ++j)
      Yb[(size_t)(row0 + hi * 8 + j) * kN + t * 16 + laneM] = f2bf(acc[t][j] + bv[j]);
}

// ---------------------------------------------------------------------------
// conv3x3 (dil 1 or 2, 'same') as implicit GEMM. K chunk = 32 channels of the
// packed 8x8 image staged by the same double-buffered TDM path; the zero halo
// is realized by zeroing the B fragment for out-of-image columns.
// ---------------------------------------------------------------------------
__global__ void __launch_bounds__(128)
conv3x3(const float* __restrict__ W, const float* __restrict__ bias,
        const bfu* __restrict__ X, bfu* __restrict__ Y, int dil) {
  __shared__ __attribute__((aligned(16))) bfu xs[2 * kCH];
  const int b     = blockIdx.y;
  const int wave  = threadIdx.x >> 5;
  const int lane  = threadIdx.x & 31;
  const int laneM = lane & 15;
  const int hi    = lane >> 4;
  const int row0  = blockIdx.x * 64 + wave * 16;
  const bfu* Xb   = X + (size_t)b * kC * kN;

  v8f acc[4] = {};

#ifdef USE_TDM
  const unsigned lds0 = (unsigned)(unsigned long long)(void*)xs;
  if (wave == 0) tdm_copy(lds0, Xb, 32 * kN);
#endif
  for (int c0 = 0; c0 < kC; c0 += 32) {
    const int cur = (c0 >> 5) & 1;
    const bfu* xsc = xs + cur * kCH;
#ifdef USE_TDM
    if (wave == 0) {
      if (c0 + 32 < kC) {
        tdm_copy(lds0 + (unsigned)((1 - cur) * kCH * 2),
                 Xb + (size_t)(c0 + 32) * kN, 32 * kN);
        __builtin_amdgcn_s_wait_tensorcnt(1);
      } else {
        __builtin_amdgcn_s_wait_tensorcnt(0);
      }
    }
    __syncthreads();
#else
    __syncthreads();
    for (int i = threadIdx.x * 2; i < 32 * kN; i += 128 * 2) {
      const int kk = i >> 6, nn = i & 63;
      *(unsigned*)(xs + cur * kCH + kk * kS + nn) =
          *(const unsigned*)(Xb + (size_t)c0 * kN + i);
    }
    __syncthreads();
#endif
#pragma unroll 1
    for (int tap = 0; tap < 9; ++tap) {
      const int ky = tap / 3, kx = tap % 3;
      const int dy = (ky - 1) * dil, dx = (kx - 1) * dil;
      BF16Frag a;   // K index = channel within chunk; W channel stride = 9 floats
      const float* wr = W + (size_t)(row0 + laneM) * (kC * 9) + (size_t)c0 * 9 + tap;
#pragma unroll
      for (int e = 0; e < 8; ++e) {
        a.s[e]     = f2bf(wr[(hi * 8 + e) * 9]);
        a.s[8 + e] = f2bf(wr[(16 + hi * 8 + e) * 9]);
      }
      const int klo = hi * 8;
#pragma unroll
      for (int t = 0; t < 4; ++t) {
        const int n  = t * 16 + laneM;
        const int yy = (n >> 3) + dy, xx = (n & 7) + dx;
        BF16Frag bm;
        if ((unsigned)yy < 8u && (unsigned)xx < 8u) {
          const bfu* col = xsc + klo * kS + yy * kW + xx;   // == n + 8*dy + dx
#pragma unroll
          for (int e = 0; e < 16; ++e) bm.s[e] = col[(e < 8 ? e : e + 8) * kS];
        } else {
#pragma unroll
          for (int e = 0; e < 16; ++e) bm.s[e] = 0;          // zero-pad halo
        }
        acc[t] = __builtin_amdgcn_wmma_f32_16x16x32_bf16(false, a.v, false, bm.v,
                                                         (short)0, acc[t], false, false);
      }
    }
    __syncthreads();
  }
  float bv[8];
#pragma unroll
  for (int j = 0; j < 8; ++j) bv[j] = bias[row0 + hi * 8 + j];
  bfu* Yb = Y + (size_t)b * kC * kN;
#pragma unroll
  for (int t = 0; t < 4; ++t)
#pragma unroll
    for (int j = 0; j < 8; ++j)
      Yb[(size_t)(row0 + hi * 8 + j) * kN + t * 16 + laneM] = f2bf(acc[t][j] + bv[j]);
}

// ---------------------------------------------------------------------------
// f32 -> bf16 tensor conversion (inputs, once per call)
// ---------------------------------------------------------------------------
__global__ void __launch_bounds__(256)
f32_to_bf16(const float* __restrict__ X, bfu* __restrict__ Y) {
  const int i = blockIdx.x * 256 + threadIdx.x;
  Y[i] = f2bf(X[i]);
}

// ---------------------------------------------------------------------------
// BatchNorm (training-mode batch stats over B,H,W) -> per-channel scale/shift
// ---------------------------------------------------------------------------
__global__ void __launch_bounds__(256)
bn_stats(const bfu* __restrict__ X, const float* __restrict__ g,
         const float* __restrict__ beta, float* __restrict__ scale,
         float* __restrict__ shift) {
  __shared__ float ssum[256], ssq[256];
  const int c = blockIdx.x;
  float s = 0.f, q = 0.f;
  for (int i = threadIdx.x; i < kB * kN; i += 256) {
    const int b = i >> 6, n = i & 63;
    const float v = bf2f(X[((size_t)b * kC + c) * kN + n]);
    s += v; q += v * v;
  }
  ssum[threadIdx.x] = s; ssq[threadIdx.x] = q;
  __syncthreads();
  for (int st = 128; st > 0; st >>= 1) {
    if (threadIdx.x < st) {
      ssum[threadIdx.x] += ssum[threadIdx.x + st];
      ssq[threadIdx.x]  += ssq[threadIdx.x + st];
    }
    __syncthreads();
  }
  if (threadIdx.x == 0) {
    const float inv = 1.f / (float)(kB * kN);
    const float m  = ssum[0] * inv;
    const float vr = ssq[0] * inv - m * m;
    const float sc = g[c] * rsqrtf(vr + 1e-5f);
    scale[c] = sc;
    shift[c] = beta[c] - m * sc;
  }
}

// y = (res? res + : ) lrelu(x*scale[c] + shift[c]); writes bf16 (Yh) or f32 (Yf)
__global__ void __launch_bounds__(256)
bn_lrelu_res(const bfu* __restrict__ X, const float* __restrict__ scale,
             const float* __restrict__ shift, const bfu* __restrict__ res,
             bfu* __restrict__ Yh, float* __restrict__ Yf) {
  const int i = blockIdx.x * 256 + threadIdx.x;
  const int c = (i >> 6) & (kC - 1);
  float v = bf2f(X[i]) * scale[c] + shift[c];
  v = v >= 0.f ? v : 0.2f * v;
  if (res) v += bf2f(res[i]);
  if (Yf) Yf[i] = v; else Yh[i] = f2bf(v);
}

// ---------------------------------------------------------------------------
// CMA: attn[b,n,m] = softmax_m( sum_c pq[b,c,n]*pk[b,c,m] ), c over 512
// ---------------------------------------------------------------------------
__global__ void __launch_bounds__(64)
cma_attn(const bfu* __restrict__ pq, const bfu* __restrict__ pk,
         float* __restrict__ attn) {
  __shared__ float qcol[kDK];
  __shared__ float red[64];
  const int b = blockIdx.y, n = blockIdx.x, m = threadIdx.x;
  const bfu* pqb = pq + (size_t)b * kDK * kN;
  const bfu* pkb = pk + (size_t)b * kDK * kN;
  for (int c = threadIdx.x; c < kDK; c += 64) qcol[c] = bf2f(pqb[c * kN + n]);
  __syncthreads();
  float e = 0.f;
  for (int c = 0; c < kDK; ++c) e += qcol[c] * bf2f(pkb[c * kN + m]);
  red[m] = e; __syncthreads();
  for (int st = 32; st > 0; st >>= 1) { if (m < st) red[m] = fmaxf(red[m], red[m + st]); __syncthreads(); }
  const float mx = red[0]; __syncthreads();
  const float ex = __expf(e - mx);
  red[m] = ex; __syncthreads();
  for (int st = 32; st > 0; st >>= 1) { if (m < st) red[m] += red[m + st]; __syncthreads(); }
  attn[((size_t)b * kN + n) * kN + m] = ex / red[0];
}

// out[b,c,n] = gamma * sum_m pv[b,c,m]*attn[b,n,m] + x[b,c,n]   (x is f32 input)
__global__ void __launch_bounds__(256)
cma_apply(const bfu* __restrict__ pv, const float* __restrict__ attn,
          const float* __restrict__ x, const float* __restrict__ gamma,
          bfu* __restrict__ out) {
  const int i  = blockIdx.x * 256 + threadIdx.x;
  const int n  = i & 63;
  const size_t bc = (size_t)(i >> 6);
  const int b  = (int)(bc >> 11);
  const bfu* pvr  = pv + bc * kN;
  const float* ar = attn + ((size_t)b * kN + n) * kN;
  float s = 0.f;
#pragma unroll 4
  for (int m = 0; m < kN; ++m) s += bf2f(pvr[m]) * ar[m];
  out[i] = f2bf(gamma[0] * s + x[i]);
}

// ---------------------------------------------------------------------------
// MHA patch branches (P = 8,4,2,1). Branch 0 degenerates to identity on V.
// ---------------------------------------------------------------------------
__global__ void __launch_bounds__(256)
mha_scores(const bfu* __restrict__ q, const bfu* __restrict__ k,
           float* __restrict__ scores, int P, int cbase) {
  __shared__ float red[256];
  const int b  = blockIdx.y;
  const int ow = kW / P, nw = ow * ow;
  const int n  = blockIdx.x / nw, m = blockIdx.x % nw;
  const int ny = (n / ow) * P, nx = (n % ow) * P;
  const int my = (m / ow) * P, mx = (m % ow) * P;
  const int pp = P * P, d = kDK * pp;
  const bfu* qb = q + ((size_t)b * kC + cbase) * kN;
  const bfu* kb = k + ((size_t)b * kC + cbase) * kN;
  float s = 0.f;
  for (int e = threadIdx.x; e < d; e += 256) {
    const int cc = e / pp, r = e % pp, py = r / P, px = r % P;
    s += bf2f(qb[cc * kN + (ny + py) * kW + nx + px]) *
         bf2f(kb[cc * kN + (my + py) * kW + mx + px]);
  }
  red[threadIdx.x] = s; __syncthreads();
  for (int st = 128; st > 0; st >>= 1) {
    if (threadIdx.x < st) red[threadIdx.x] += red[threadIdx.x + st];
    __syncthreads();
  }
  if (threadIdx.x == 0)
    scores[((size_t)b * nw + n) * nw + m] = red[0] * rsqrtf((float)d);
}

__global__ void __launch_bounds__(64)
mha_softmax(float* __restrict__ scores, int nw) {
  __shared__ float red[64];
  const int b = blockIdx.y, n = blockIdx.x, m = threadIdx.x;
  float* row = scores + ((size_t)b * nw + n) * nw;
  const float e = (m < nw) ? row[m] : -1e30f;
  red[m] = e; __syncthreads();
  for (int st = 32; st > 0; st >>= 1) { if (m < st) red[m] = fmaxf(red[m], red[m + st]); __syncthreads(); }
  const float mx = red[0]; __syncthreads();
  const float ex = (m < nw) ? __expf(e - mx) : 0.f;
  red[m] = ex; __syncthreads();
  for (int st = 32; st > 0; st >>= 1) { if (m < st) red[m] += red[m + st]; __syncthreads(); }
  if (m < nw) row[m] = ex / red[0];
}

__global__ void __launch_bounds__(256)
mha_apply(const bfu* __restrict__ v, const float* __restrict__ attn,
          bfu* __restrict__ y, int P, int cbase) {
  const int i  = blockIdx.x * 256 + threadIdx.x;   // over B*512*64
  const int hw = i & 63;
  const int bc = i >> 6;
  const int cc = bc & (kDK - 1);
  const int b  = bc >> 9;
  const int h = hw >> 3, w = hw & 7;
  const int ow = kW / P, nw = ow * ow;
  const int n = (h / P) * ow + (w / P);
  const int py = h % P, px = w % P;
  const float* ar = attn + ((size_t)b * nw + n) * nw;
  const bfu* vb = v + ((size_t)b * kC + cbase + cc) * kN;
  float s = 0.f;
  for (int m = 0; m < nw; ++m) {
    const int my = (m / ow) * P + py, mx = (m % ow) * P + px;
    s += ar[m] * bf2f(vb[my * kW + mx]);
  }
  y[((size_t)b * kC + cbase + cc) * kN + hw] = f2bf(s);
}

// ---------------------------------------------------------------------------
// Host orchestration. Input order (JAX pytree, nested dicts key-sorted):
//   0 rgb_feat, 1 freq_feat,
//   blocks (2x, 18 leaves): b1,bk,bo,bq,bv,f1b,f1w,f2b,f2w,fb1,fb2,fg1,fg2,g1,wk,wo,wq,wv
//   cma (7 leaves): bk,bq,bv,gamma,wk,wq,wv
// ---------------------------------------------------------------------------
extern "C" void kernel_launch(void* const* d_in, const int* in_sizes, int n_in,
                              void* d_out, int out_size, void* d_ws, size_t ws_size,
                              hipStream_t stream) {
  (void)in_sizes; (void)n_in; (void)out_size; (void)ws_size;
  auto F = [&](int i) { return (const float*)d_in[i]; };

  const float* rgb  = F(0);
  const float* freq = F(1);

  const int CMA = 2 + 2 * 18;
  const float* cma_bk = F(CMA + 0); const float* cma_bq = F(CMA + 1);
  const float* cma_bv = F(CMA + 2); const float* cma_g  = F(CMA + 3);
  const float* cma_wk = F(CMA + 4); const float* cma_wq = F(CMA + 5);
  const float* cma_wv = F(CMA + 6);

  char* wsb = (char*)d_ws;
  const size_t ACTH = (size_t)kB * kC * kN * sizeof(bfu);    // 8 MiB (bf16 tensor)
  bfu* xcur  = (bfu*)(wsb + 0 * ACTH);
  bfu* q     = (bfu*)(wsb + 1 * ACTH);
  bfu* k     = (bfu*)(wsb + 2 * ACTH);
  bfu* v     = (bfu*)(wsb + 3 * ACTH);
  bfu* tA    = (bfu*)(wsb + 4 * ACTH);
  bfu* tB    = (bfu*)(wsb + 5 * ACTH);
  bfu* out1  = (bfu*)(wsb + 6 * ACTH);
  bfu* rgbh  = (bfu*)(wsb + 7 * ACTH);
  bfu* freqh = (bfu*)(wsb + 8 * ACTH);
  float* scale   = (float*)(wsb + 9 * ACTH);
  float* shift   = scale + kC;
  float* attnbuf = shift + kC;                  // up to 32*64*64 f32

  const dim3 g512(kDK / 64, kB), g2048(kC / 64, kB);
  const int TOT   = kB * kC * kN;               // 4,194,304
  const int TOT_B = TOT / 256;

  // inputs -> bf16 (once)
  f32_to_bf16<<<TOT_B, 256, 0, stream>>>(rgb,  rgbh);
  f32_to_bf16<<<TOT_B, 256, 0, stream>>>(freq, freqh);

  // ---------------- CMA ----------------
  gemm1x1<<<g512,  128, 0, stream>>>(cma_wq, cma_bq, rgbh,  q, kDK);
  gemm1x1<<<g512,  128, 0, stream>>>(cma_wk, cma_bk, freqh, k, kDK);
  gemm1x1<<<g2048, 128, 0, stream>>>(cma_wv, cma_bv, freqh, v, kC);
  cma_attn<<<dim3(kN, kB), 64, 0, stream>>>(q, k, attnbuf);
  cma_apply<<<TOT_B, 256, 0, stream>>>(v, attnbuf, rgb, cma_g, xcur);

  // ---------------- Transformer blocks ----------------
  for (int bi = 0; bi < 2; ++bi) {
    const int base = 2 + bi * 18;
    const float* b1  = F(base + 0);  const float* bk  = F(base + 1);
    const float* bo  = F(base + 2);  const float* bq  = F(base + 3);
    const float* bvp = F(base + 4);  const float* f1b = F(base + 5);
    const float* f1w = F(base + 6);  const float* f2b = F(base + 7);
    const float* f2w = F(base + 8);  const float* fb1 = F(base + 9);
    const float* fb2 = F(base + 10); const float* fg1 = F(base + 11);
    const float* fg2 = F(base + 12); const float* g1  = F(base + 13);
    const float* wk  = F(base + 14); const float* wo  = F(base + 15);
    const float* wq  = F(base + 16); const float* wv  = F(base + 17);

    gemm1x1<<<g2048, 128, 0, stream>>>(wq, bq,  xcur, q, kC);
    gemm1x1<<<g2048, 128, 0, stream>>>(wk, bk,  xcur, k, kC);
    gemm1x1<<<g2048, 128, 0, stream>>>(wv, bvp, xcur, v, kC);

    for (int br = 0; br < 4; ++br) {
      const int P = 8 >> br;                 // 8,4,2,1
      const int ow = kW / P, nw = ow * ow;   // 1,4,16,64
      const int cbase = br * kDK;
      mha_scores <<<dim3(nw * nw, kB), 256, 0, stream>>>(q, k, attnbuf, P, cbase);
      mha_softmax<<<dim3(nw, kB),       64, 0, stream>>>(attnbuf, nw);
      mha_apply  <<<(kB * kDK * kN) / 256, 256, 0, stream>>>(v, attnbuf, tA, P, cbase);
    }

    // out1 = x + lrelu(bn(conv3x3(tA)))
    conv3x3<<<g2048, 128, 0, stream>>>(wo, bo, tA, tB, 1);
    bn_stats<<<kC, 256, 0, stream>>>(tB, g1, b1, scale, shift);
    bn_lrelu_res<<<TOT_B, 256, 0, stream>>>(tB, scale, shift, xcur, out1, nullptr);

    // FFN: h1 = lrelu(bn(conv3x3_dil2(out1)))  (reuse q as h1)
    conv3x3<<<g2048, 128, 0, stream>>>(f1w, f1b, out1, tA, 2);
    bn_stats<<<kC, 256, 0, stream>>>(tA, fg1, fb1, scale, shift);
    bn_lrelu_res<<<TOT_B, 256, 0, stream>>>(tA, scale, shift, nullptr, q, nullptr);

    // h2 = lrelu(bn(conv3x3(h1)));  x_next = out1 + h2
    conv3x3<<<g2048, 128, 0, stream>>>(f2w, f2b, q, tB, 1);
    bn_stats<<<kC, 256, 0, stream>>>(tB, fg2, fb2, scale, shift);
    if (bi == 1)
      bn_lrelu_res<<<TOT_B, 256, 0, stream>>>(tB, scale, shift, out1, nullptr, (float*)d_out);
    else
      bn_lrelu_res<<<TOT_B, 256, 0, stream>>>(tB, scale, shift, out1, xcur, nullptr);
  }
}
